// SeqAttention_53876069761432
// MI455X (gfx1250) — compile-verified, matrix operand order checked by reading, not run on hardware
//
#include <hip/hip_runtime.h>
#include <hip/hip_bf16.h>

// ---------------------------------------------------------------------------
// Problem constants
// ---------------------------------------------------------------------------
#define T_LEN 512
#define BATCH 64
#define D_IN  1024
#define HID   512
#define D_OUT 10
#define TB    (T_LEN * BATCH)          // 32768 GEMM rows
#define G3    (3 * HID)                // 1536

typedef __attribute__((ext_vector_type(16))) _Float16 v16h;
typedef __attribute__((ext_vector_type(8)))  float    v8f;

union V16HU { v16h v; unsigned u[8]; uint4 q[2]; _Float16 h[16]; };

// CDNA5 16-bit 16x32 tile, per-lane layout (lane = row, K across regs):
//   regs v0..3 : K = 8*half + {0..7}   -> one contiguous 16B chunk
//   regs v4..7 : K = 16 + 8*half + {0..7} -> second contiguous 16B chunk
// so each operand is exactly two b128 loads per lane.

// Load a 16(rows) x 32(K) f16 tile from an LDS row-major buffer.
// Used for both A (row = M) and B (row = N); layouts are symmetric.
__device__ __forceinline__ v16h load_tile_lds(const _Float16* lds, int stride,
                                              int r0, int c0) {
  int lane = threadIdx.x & 31;
  int row  = r0 + (lane & 15);
  int half = lane >> 4;
  const _Float16* p = lds + row * stride + c0 + (half << 3);
  V16HU o;
  o.q[0] = *(const uint4*)p;          // ds_load_b128
  o.q[1] = *(const uint4*)(p + 16);   // ds_load_b128
  return o.v;
}

// Load a 32(K) x 16(N) f16 B tile from global W stored (N,K) row-major.
__device__ __forceinline__ v16h load_b_nk(const _Float16* __restrict__ W,
                                          int ldk, int n0, int k0) {
  int lane = threadIdx.x & 31;
  int col  = n0 + (lane & 15);
  int half = lane >> 4;
  const _Float16* p = W + (size_t)col * ldk + k0 + (half << 3);
  V16HU o;
  o.q[0] = *(const uint4*)p;          // global_load_b128
  o.q[1] = *(const uint4*)(p + 16);   // global_load_b128
  return o.v;
}

__device__ __forceinline__ float sigmoidf_(float x) {
  return 1.0f / (1.0f + __expf(-x));
}

// ---------------------------------------------------------------------------
// Conversion kernels (one-time precision drop to f16)
// ---------------------------------------------------------------------------
__global__ void cvt_f32_f16(const float* __restrict__ src,
                            _Float16* __restrict__ dst, long n) {
  long i = (long)blockIdx.x * blockDim.x + threadIdx.x;
  long stride = (long)gridDim.x * blockDim.x;
  for (; i < n; i += stride) dst[i] = (_Float16)src[i];
}

// src is (K,N) row-major f32 -> dst (N,K) row-major f16
__global__ void cvt_f32_f16_T(const float* __restrict__ src,
                              _Float16* __restrict__ dst, int K, int N) {
  long i = (long)blockIdx.x * blockDim.x + threadIdx.x;
  long total = (long)K * N;
  long stride = (long)gridDim.x * blockDim.x;
  for (; i < total; i += stride) {
    int n = (int)(i / K), k = (int)(i % K);
    dst[i] = (_Float16)src[(size_t)k * N + n];
  }
}

__global__ void init_kernel(float* __restrict__ scores, unsigned* __restrict__ bar) {
  int i = blockIdx.x * blockDim.x + threadIdx.x;
  if (i < TB) scores[i] = 0.0f;
  if (i < 32) bar[i] = 0u;
}

// ---------------------------------------------------------------------------
// Generic WMMA GEMM: C = A(MxK f16,row-major) * B^T (B is (N,K) f16) + bias
// Block = 256 thr (8 waves), block tile 64M x 128N.
// A staged through double-buffered LDS (one barrier per k-chunk); all 4 B
// tiles fetched (b128) before the 4 WMMAs so loads overlap the matrix pipe.
// MODE 0: store f16 (bias)            -> xp projection
// MODE 1: fused tanh(.+bias)*ctxw, half-wave reduce, atomicAdd scores[row]
// MODE 2: store f16 relu(.+bias)      -> classifier layer 1
// ---------------------------------------------------------------------------
template <int MODE>
__global__ void wmma_gemm_kernel(const _Float16* __restrict__ A,
                                 const _Float16* __restrict__ B,
                                 const float* __restrict__ bias,
                                 _Float16* __restrict__ Cout,
                                 float* __restrict__ scores,
                                 const float* __restrict__ ctxw,
                                 int N, int K) {
  __shared__ _Float16 As[2][64 * 32];
  int tid = threadIdx.x, lane = tid & 31, wave = tid >> 5;
  int mt = wave & 3;        // 4 M-tiles of 16
  int nq = wave >> 2;       // 2 N-quadrants of 64
  int rowBase = blockIdx.y * 64;
  int nBase   = blockIdx.x * 128;
  int sr = tid >> 2, sq = tid & 3;          // staging: 64 rows x 4 xb128

  v8f acc[4] = {v8f{}, v8f{}, v8f{}, v8f{}};
  int kIters = K >> 5;
  // prologue: stage k-chunk 0
  *(uint4*)(As[0] + sr * 32 + sq * 8) =
      *(const uint4*)(A + (size_t)(rowBase + sr) * K + sq * 8);
  __syncthreads();
  int buf = 0;
  for (int kc = 0; kc < kIters; ++kc) {
    if (kc + 1 < kIters) {  // stage next chunk into the other buffer
      *(uint4*)(As[buf ^ 1] + sr * 32 + sq * 8) =
          *(const uint4*)(A + (size_t)(rowBase + sr) * K + (kc + 1) * 32 + sq * 8);
    }
    v16h a  = load_tile_lds(As[buf], 32, mt * 16, 0);
    v16h b0 = load_b_nk(B, K, nBase + nq * 64 + 0,  kc * 32);
    v16h b1 = load_b_nk(B, K, nBase + nq * 64 + 16, kc * 32);
    v16h b2 = load_b_nk(B, K, nBase + nq * 64 + 32, kc * 32);
    v16h b3 = load_b_nk(B, K, nBase + nq * 64 + 48, kc * 32);
    acc[0] = __builtin_amdgcn_wmma_f32_16x16x32_f16(false, a, false, b0,
                                                    (short)0, acc[0], false, false);
    acc[1] = __builtin_amdgcn_wmma_f32_16x16x32_f16(false, a, false, b1,
                                                    (short)0, acc[1], false, false);
    acc[2] = __builtin_amdgcn_wmma_f32_16x16x32_f16(false, a, false, b2,
                                                    (short)0, acc[2], false, false);
    acc[3] = __builtin_amdgcn_wmma_f32_16x16x32_f16(false, a, false, b3,
                                                    (short)0, acc[3], false, false);
    __syncthreads();
    buf ^= 1;
  }

  int colLane = lane & 15;
  int rBase   = rowBase + mt * 16 + (lane >> 4) * 8;
  if (MODE == 1) {
    float s[8] = {0, 0, 0, 0, 0, 0, 0, 0};
#pragma unroll
    for (int j = 0; j < 4; ++j) {
      int n = nBase + nq * 64 + j * 16 + colLane;
      float bv = bias[n], cw = ctxw[n];
#pragma unroll
      for (int r = 0; r < 8; ++r) s[r] += tanhf(acc[j][r] + bv) * cw;
    }
#pragma unroll
    for (int r = 0; r < 8; ++r) {
      float v = s[r];
      for (int off = 1; off < 16; off <<= 1) v += __shfl_xor(v, off, 32);
      if (colLane == 0) atomicAdd(&scores[rBase + r], v);
    }
  } else {
#pragma unroll
    for (int j = 0; j < 4; ++j) {
      int n = nBase + nq * 64 + j * 16 + colLane;
      float bv = bias[n];
#pragma unroll
      for (int r = 0; r < 8; ++r) {
        float v = acc[j][r] + bv;
        if (MODE == 2) v = fmaxf(v, 0.0f);
        Cout[(size_t)(rBase + r) * N + n] = (_Float16)v;
      }
    }
  }
}

// ---------------------------------------------------------------------------
// Persistent bidirectional GRU recurrence.
// 32 blocks: blocks 0..15 forward, 16..31 backward; block owns 32 h-columns.
//
// LDS (dynamic, 188416 B -> one block per WGP, within the 320KB limit):
//   hPrev : 64 x 512 f16  (64 KB)   async-staged every step (ASYNCcnt)
//   whhS  : 96 x 512 f16  (96 KB)   this block's w_hh slab, staged ONCE
//   gh    : 64 x 96  f32  (24 KB)   gate pre-activations
//
// Per step: 8 waves x 3 WMMA tiles x 16 k-steps (software-pipelined
// ds_load_b128 -> v_wmma), elementwise GRU update, then a device-scope
// sense barrier among the direction's 16 blocks (all 32 blocks resident).
// ---------------------------------------------------------------------------
__global__ void gru_kernel(const _Float16* __restrict__ xpF,
                           const _Float16* __restrict__ xpB,
                           const _Float16* __restrict__ whhF,
                           const _Float16* __restrict__ whhB,
                           const float* __restrict__ bhhF,
                           const float* __restrict__ bhhB,
                           _Float16* __restrict__ fout,
                           unsigned* __restrict__ barriers) {
  extern __shared__ char smem[];
  _Float16* hPrev = (_Float16*)smem;                         // 64 x 512
  _Float16* whhS  = (_Float16*)(smem + 64 * 512 * 2);        // 96 x 512
  float*    gh    = (float*)(smem + (64 + 96) * 512 * 2);    // 64 x 96

  int dir = blockIdx.x >> 4;
  int blk = blockIdx.x & 15;
  const _Float16* xp  = dir ? xpB  : xpF;
  const _Float16* whh = dir ? whhB : whhF;
  const float*    bhh = dir ? bhhB : bhhF;
  unsigned* bar = barriers + dir * 16;                // [0]=count [1]=gen
  int j0 = blk * 32;
  int dirOff = dir ? HID : 0;
  int tid = threadIdx.x, lane = tid & 31, wave = tid >> 5;

  // ---- stage this block's w_hh slab into LDS once (96 rows x 512 K) ----
  // slab row r: gate g = r/32, w_hh row = g*HID + j0 + (r%32)
  for (int i = tid; i < 96 * 64; i += 256) {          // 64 uint4 per row
    int r = i >> 6, c = i & 63;
    int g = r >> 5;
    ((uint4*)whhS)[i] = *(const uint4*)(
        whh + (size_t)(g * HID + j0 + (r & 31)) * 512 + c * 8);
  }

  for (int s = 0; s < T_LEN; ++s) {
    int ta = dir ? (T_LEN - 1 - s) : s;               // actual time index
    int tp = dir ? (ta + 1) : (ta - 1);               // previous actual time
    // prefetch this step's xp activations while WMMA phase runs
    {
      int b = tid >> 2, q = tid & 3;                  // 64 rows x 4 lines
      __builtin_prefetch(xp + ((size_t)ta * BATCH + b) * G3 + j0 + q * 8, 0, 1);
    }
    // ---- stage h_{prev} into LDS ----
    if (s == 0) {
      uint4 z = {0, 0, 0, 0};
      for (int i = tid; i < (64 * 512) / 8; i += 256) ((uint4*)hPrev)[i] = z;
    } else {
      // async global->LDS copy, GVS mode: SGPR base + per-lane 32-bit offset
#pragma unroll
      for (int i = tid; i < 4096; i += 256) {         // 64 uint4 per row
        int b = i >> 6, c = i & 63;
        unsigned ldsDst = (unsigned)(unsigned long long)(hPrev + b * 512 + c * 8);
        unsigned srcOff = (unsigned)(
            (((size_t)tp * BATCH + b) * (2 * HID) + dirOff + c * 8) * sizeof(_Float16));
        asm volatile("global_load_async_to_lds_b128 %0, %1, %2"
                     :: "v"(ldsDst), "v"(srcOff), "s"(fout) : "memory");
      }
      asm volatile("s_wait_asynccnt 0x0" ::: "memory");
    }
    __syncthreads();
    // ---- gh = h_prev @ whh^T + bhh for this block's 3x(64x32) gate slabs ----
#pragma unroll
    for (int i = 0; i < 3; ++i) {
      int tile = wave * 3 + i;                        // 0..23
      int g = tile >> 3, rem = tile & 7;
      int mt = rem >> 1, nt = rem & 1;
      int nLoc = g * 32 + nt * 16;                    // row in whhS slab
      // software-pipelined k-loop: loads for k+1 issue under WMMA k
      v16h a = load_tile_lds(hPrev, 512, mt * 16, 0);
      v16h b = load_tile_lds(whhS, 512, nLoc, 0);
      v8f acc = {};
#pragma unroll
      for (int kc = 0; kc < 16; ++kc) {
        v16h an = a, bn = b;
        if (kc < 15) {
          an = load_tile_lds(hPrev, 512, mt * 16, (kc + 1) * 32);
          bn = load_tile_lds(whhS, 512, nLoc, (kc + 1) * 32);
        }
        acc = __builtin_amdgcn_wmma_f32_16x16x32_f16(
            false, a, false, b, (short)0, acc, false, false);
        a = an; b = bn;
      }
      float bv = bhh[g * HID + j0 + nt * 16 + (lane & 15)];
      int colg = nLoc + (lane & 15);
      int rb   = mt * 16 + (lane >> 4) * 8;
#pragma unroll
      for (int r = 0; r < 8; ++r) gh[(rb + r) * 96 + colg] = acc[r] + bv;
    }
    __syncthreads();
    // ---- elementwise GRU update, 8 elements/thread ----
#pragma unroll
    for (int e = 0; e < 8; ++e) {
      int id = tid + e * 256;                         // 0..2047
      int b = id >> 5, jc = id & 31;
      size_t xrow = ((size_t)ta * BATCH + b) * G3 + j0 + jc;
      float xr = (float)xp[xrow];
      float xz = (float)xp[xrow + HID];
      float xn = (float)xp[xrow + 2 * HID];
      float r_ = sigmoidf_(xr + gh[b * 96 + jc]);
      float z_ = sigmoidf_(xz + gh[b * 96 + 32 + jc]);
      float n_ = tanhf(xn + r_ * gh[b * 96 + 64 + jc]);
      float hp = (float)hPrev[b * 512 + j0 + jc];
      float hn = (1.0f - z_) * n_ + z_ * hp;
      fout[((size_t)ta * BATCH + b) * (2 * HID) + dirOff + j0 + jc] = (_Float16)hn;
    }
    // ---- device-scope barrier over this direction's 16 blocks ----
    __threadfence();
    __syncthreads();
    if (tid == 0) {
      unsigned gen = atomicAdd(&bar[1], 0u);
      unsigned arrived = atomicAdd(&bar[0], 1u);
      if (arrived == 15u) {
        atomicExch(&bar[0], 0u);
        atomicAdd(&bar[1], 1u);
      } else {
        while (atomicAdd(&bar[1], 0u) == gen) __builtin_amdgcn_s_sleep(2);
      }
    }
    __syncthreads();
    __threadfence();
  }
}

// ---------------------------------------------------------------------------
// scores -> tanh -> softmax over T per batch row
// ---------------------------------------------------------------------------
__global__ void softmax_kernel(const float* __restrict__ scores,
                               float* __restrict__ w) {
  __shared__ float red[T_LEN];
  int b = blockIdx.x, t = threadIdx.x;
  float s = tanhf(scores[(size_t)t * BATCH + b]);
  red[t] = s;
  __syncthreads();
  for (int off = T_LEN / 2; off > 0; off >>= 1) {
    if (t < off) red[t] = fmaxf(red[t], red[t + off]);
    __syncthreads();
  }
  float mx = red[0];
  __syncthreads();
  float e = __expf(s - mx);
  red[t] = e;
  __syncthreads();
  for (int off = T_LEN / 2; off > 0; off >>= 1) {
    if (t < off) red[t] += red[t + off];
    __syncthreads();
  }
  w[(size_t)b * T_LEN + t] = e / red[0];
}

// ctx[b,d] = sum_t fout[t,b,d] * w[b,t]
__global__ void ctx_kernel(const _Float16* __restrict__ fout,
                           const float* __restrict__ w,
                           _Float16* __restrict__ ctx) {
  int b = blockIdx.x;
  int d = blockIdx.y * 128 + threadIdx.x;
  float acc = 0.0f;
  for (int t = 0; t < T_LEN; ++t)
    acc += (float)fout[((size_t)t * BATCH + b) * (2 * HID) + d] *
           w[(size_t)b * T_LEN + t];
  ctx[(size_t)b * (2 * HID) + d] = (_Float16)acc;
}

// out[b,o] = hid[b,:] . cls_w2[:,o] + b2[o]
__global__ void cls2_kernel(const _Float16* __restrict__ hid,
                            const float* __restrict__ w2,
                            const float* __restrict__ b2,
                            float* __restrict__ out) {
  int i = blockIdx.x * 64 + threadIdx.x;
  if (i >= BATCH * D_OUT) return;
  int b = i / D_OUT, o = i % D_OUT;
  float acc = b2[o];
  for (int h = 0; h < HID; ++h)
    acc += (float)hid[(size_t)b * HID + h] * w2[(size_t)h * D_OUT + o];
  out[i] = acc;
}

// ---------------------------------------------------------------------------
// Host orchestration
// ---------------------------------------------------------------------------
extern "C" void kernel_launch(void* const* d_in, const int* in_sizes, int n_in,
                              void* d_out, int out_size, void* d_ws, size_t ws_size,
                              hipStream_t stream) {
  (void)in_sizes; (void)n_in; (void)out_size; (void)ws_size;
  // inputs (setup_inputs order); d_in[0] 'numerical' is unused by reference
  const float* embedding = (const float*)d_in[1];
  const float* w_ih_f = (const float*)d_in[2];
  const float* w_hh_f = (const float*)d_in[3];
  const float* b_ih_f = (const float*)d_in[4];
  const float* b_hh_f = (const float*)d_in[5];
  const float* w_ih_b = (const float*)d_in[6];
  const float* w_hh_b = (const float*)d_in[7];
  const float* b_ih_b = (const float*)d_in[8];
  const float* b_hh_b = (const float*)d_in[9];
  const float* attn_w = (const float*)d_in[10];
  const float* attn_b = (const float*)d_in[11];
  const float* ctx_w  = (const float*)d_in[12];
  const float* cls_w1 = (const float*)d_in[13];
  const float* cls_b1 = (const float*)d_in[14];
  const float* cls_w2 = (const float*)d_in[15];
  const float* cls_b2 = (const float*)d_in[16];
  float* out = (float*)d_out;

  // workspace carve-up (256B aligned)
  size_t off = 0;
  auto alloc = [&](size_t bytes) -> void* {
    off = (off + 255) & ~(size_t)255;
    void* p = (char*)d_ws + off;
    off += bytes;
    return p;
  };
  unsigned*  bar   = (unsigned*)alloc(128);
  float*     scor  = (float*)alloc((size_t)TB * 4);
  _Float16*  embH  = (_Float16*)alloc((size_t)TB * D_IN * 2);
  _Float16*  wihF  = (_Float16*)alloc((size_t)G3 * D_IN * 2);
  _Float16*  wihB  = (_Float16*)alloc((size_t)G3 * D_IN * 2);
  _Float16*  whhF  = (_Float16*)alloc((size_t)G3 * HID * 2);
  _Float16*  whhB  = (_Float16*)alloc((size_t)G3 * HID * 2);
  _Float16*  attnT = (_Float16*)alloc((size_t)(2 * HID) * (2 * HID) * 2); // (N,K)
  _Float16*  w1T   = (_Float16*)alloc((size_t)HID * D_IN * 2);            // (N,K)
  _Float16*  xpF   = (_Float16*)alloc((size_t)TB * G3 * 2);
  _Float16*  xpB   = (_Float16*)alloc((size_t)TB * G3 * 2);
  _Float16*  fout  = (_Float16*)alloc((size_t)TB * 2 * HID * 2);
  float*     wts   = (float*)alloc((size_t)BATCH * T_LEN * 4);
  _Float16*  ctxH  = (_Float16*)alloc((size_t)BATCH * 2 * HID * 2);
  _Float16*  hidH  = (_Float16*)alloc((size_t)BATCH * HID * 2);

  // 1) precision conversions
  cvt_f32_f16<<<8192, 256, 0, stream>>>(embedding, embH, (long)TB * D_IN);
  cvt_f32_f16<<<1024, 256, 0, stream>>>(w_ih_f, wihF, (long)G3 * D_IN);
  cvt_f32_f16<<<1024, 256, 0, stream>>>(w_ih_b, wihB, (long)G3 * D_IN);
  cvt_f32_f16<<<512,  256, 0, stream>>>(w_hh_f, whhF, (long)G3 * HID);
  cvt_f32_f16<<<512,  256, 0, stream>>>(w_hh_b, whhB, (long)G3 * HID);
  cvt_f32_f16_T<<<1024, 256, 0, stream>>>(attn_w, attnT, 2 * HID, 2 * HID);
  cvt_f32_f16_T<<<512,  256, 0, stream>>>(cls_w1, w1T, D_IN, HID);

  // 2) input projections xp = emb @ w_ih^T + b_ih  (both directions)
  wmma_gemm_kernel<0><<<dim3(G3 / 128, TB / 64), 256, 0, stream>>>(
      embH, wihF, b_ih_f, xpF, nullptr, nullptr, G3, D_IN);
  wmma_gemm_kernel<0><<<dim3(G3 / 128, TB / 64), 256, 0, stream>>>(
      embH, wihB, b_ih_b, xpB, nullptr, nullptr, G3, D_IN);

  // 3) zero barrier + fused-score accumulator
  init_kernel<<<128, 256, 0, stream>>>(scor, bar);

  // 4) persistent bidirectional recurrence
  //    dynamic LDS: (64 + 96)*512*2 + 64*96*4 = 188416 B  (one block/WGP)
  gru_kernel<<<32, 256, (64 + 96) * 512 * 2 + 64 * 96 * 4, stream>>>(
      xpF, xpB, whhF, whhB, b_hh_f, b_hh_b, fout, bar);

  // 5) attention: tanh(fout @ attn_w + attn_b) . ctx_w fused into scores
  wmma_gemm_kernel<1><<<dim3((2 * HID) / 128, TB / 64), 256, 0, stream>>>(
      fout, attnT, attn_b, nullptr, scor, ctx_w, 2 * HID, 2 * HID);

  // 6) tanh + softmax over T, then context vector
  softmax_kernel<<<BATCH, T_LEN, 0, stream>>>(scor, wts);
  ctx_kernel<<<dim3(BATCH, (2 * HID) / 128), 128, 0, stream>>>(fout, wts, ctxH);

  // 7) classifier
  wmma_gemm_kernel<2><<<dim3(HID / 128, 1), 256, 0, stream>>>(
      ctxH, w1T, cls_b1, hidH, nullptr, nullptr, HID, D_IN);
  cls2_kernel<<<(BATCH * D_OUT + 63) / 64, 64, 0, stream>>>(
      hidH, cls_w2, cls_b2, out);
}
